// MultiLayerGRU_75050258530399
// MI455X (gfx1250) — compile-verified, matrix operand order checked by reading, not run on hardware
//
#include <hip/hip_runtime.h>
#include <hip/hip_bf16.h>
#include <math.h>

typedef __bf16 bf16;
typedef __attribute__((ext_vector_type(16))) __bf16 v16bf;
typedef __attribute__((ext_vector_type(8)))  __bf16 v8bf;
typedef __attribute__((ext_vector_type(8)))  float  v8f;

#define GRU_H 512
#define GRU_B 64
#define GRU_T 1024
#define GRU_L 2
#define NG    1536   // 3*H gate columns [r | z | h]
#define KC    1024   // concat K (x part + h part) for gates GEMM
#define NBLK  48     // persistent grid: 48 blocks * 8 waves = 384 tiles
#define LDS_BYTES (131072 + 65536)   // 4x32KB gate chunks + 4x16KB candidate chunks

// ---- workspace layout (bytes) ----
#define OFF_X0 ((size_t)0)                                    // input  -> bf16 (b,t,k)
#define OFF_X1 (OFF_X0 + (size_t)GRU_B*GRU_T*GRU_H*2)         // layer0 out bf16 (t,b,k)
#define OFF_WG (OFF_X1 + (size_t)GRU_B*GRU_T*GRU_H*2)         // packed [Wx;Wh] per layer
#define OFF_WH (OFF_WG + (size_t)GRU_L*KC*NG*2)               // packed Whh per layer
#define OFF_G  (OFF_WH + (size_t)GRU_L*GRU_H*GRU_H*2)         // gates preact f32 64x1536 (z,h blocks)
#define OFF_RH (OFF_G  + (size_t)GRU_B*NG*4)                  // (r*h) bf16 64x512
#define OFF_HF (OFF_RH + (size_t)GRU_B*GRU_H*2)               // h f32 [l][b][n]
#define OFF_HB (OFF_HF + (size_t)GRU_L*GRU_B*GRU_H*4)         // h bf16 [l][b][n]
#define OFF_SY (OFF_HB + (size_t)GRU_L*GRU_B*GRU_H*2)         // sync counters

__device__ __forceinline__ float sigmoidf_(float x) { return 1.0f / (1.0f + expf(-x)); }

// ---------------- prologue kernels ----------------

__global__ void gru_cvt_in(const float* __restrict__ in, bf16* __restrict__ o, int n) {
  for (int i = blockIdx.x * blockDim.x + threadIdx.x; i < n; i += gridDim.x * blockDim.x)
    o[i] = (bf16)in[i];
}

__global__ void gru_init_h(const float* __restrict__ hidden,
                           float* __restrict__ hf, bf16* __restrict__ hb) {
  int idx = blockIdx.x * blockDim.x + threadIdx.x;        // 65536 = L*B*H
  int n = idx & (GRU_H - 1);
  int b = (idx >> 9) & (GRU_B - 1);
  int l = idx >> 15;
  float v = hidden[((size_t)b * GRU_L + l) * GRU_H + n];  // hidden is (B,L,H)
  hf[idx] = v;
  hb[idx] = (bf16)v;
}

__global__ void gru_zero_sync(unsigned* __restrict__ sy) {
  if (threadIdx.x < 64) sy[threadIdx.x] = 0u;
}

// Pack [Wx ; Wh] (1024 x 1536) per layer into WMMA B-fragment order:
// offset = ((((ntile*32 + kblk)*2 + khalf)*16 + nlocal)*16 + j), 16 contiguous bf16 per
// lane; k = kblk*32 + (j/8)*16 + khalf*8 + (j%8)  (16-bit A/B fragment layout, ISA 7.12.2)
__global__ void gru_pack_wg(const float* __restrict__ Wxr, const float* __restrict__ Whr,
                            const float* __restrict__ Wxz, const float* __restrict__ Whz,
                            const float* __restrict__ Wxh, bf16* __restrict__ wg) {
  int idx = blockIdx.x * blockDim.x + threadIdx.x;        // 2*1024*1536
  int layer = idx / (KC * NG);
  int r = idx % (KC * NG);
  int j      =  r        & 15;
  int nlocal = (r >> 4)  & 15;
  int khalf  = (r >> 8)  & 1;
  int kblk   = (r >> 9)  & 31;
  int ntile  =  r >> 14;                                  // 0..95
  int n = ntile * 16 + nlocal;
  int k = kblk * 32 + (j >> 3) * 16 + khalf * 8 + (j & 7);
  size_t wb = (size_t)layer * GRU_H * GRU_H;
  float v;
  if (k < GRU_H) {           // x-weight rows
    if      (n <     GRU_H) v = Wxr[wb + (size_t)k * GRU_H + n];
    else if (n < 2 * GRU_H) v = Wxz[wb + (size_t)k * GRU_H + (n - GRU_H)];
    else                    v = Wxh[wb + (size_t)k * GRU_H + (n - 2 * GRU_H)];
  } else {                   // h-weight rows (candidate block zero: (r*h)@Whh handled later)
    int kh = k - GRU_H;
    if      (n <     GRU_H) v = Whr[wb + (size_t)kh * GRU_H + n];
    else if (n < 2 * GRU_H) v = Whz[wb + (size_t)kh * GRU_H + (n - GRU_H)];
    else                    v = 0.0f;
  }
  wg[idx] = (bf16)v;
}

__global__ void gru_pack_wh(const float* __restrict__ Whh, bf16* __restrict__ wh) {
  int idx = blockIdx.x * blockDim.x + threadIdx.x;        // 2*512*512
  int layer = idx >> 18;
  int r = idx & ((1 << 18) - 1);
  int j      =  r        & 15;
  int nlocal = (r >> 4)  & 15;
  int khalf  = (r >> 8)  & 1;
  int kblk   = (r >> 9)  & 15;
  int ntile  =  r >> 13;                                  // 0..31
  int n = ntile * 16 + nlocal;
  int k = kblk * 32 + (j >> 3) * 16 + khalf * 8 + (j & 7);
  wh[idx] = (bf16)Whh[(size_t)layer * GRU_H * GRU_H + (size_t)k * GRU_H + n];
}

// ---------------- persistent recurrence kernel ----------------

__device__ __forceinline__ void grid_sync(unsigned* cnt, unsigned* gen) {
  __syncthreads();
  if (threadIdx.x == 0) {
    __threadfence();                                        // release prior global writes
    unsigned g = __atomic_load_n(gen, __ATOMIC_RELAXED);    // read generation BEFORE arriving
    if (atomicAdd(cnt, 1u) == NBLK - 1) {
      __atomic_store_n(cnt, 0u, __ATOMIC_RELAXED);
      __threadfence();
      __atomic_fetch_add(gen, 1u, __ATOMIC_RELEASE);
    } else {
      while (__atomic_load_n(gen, __ATOMIC_ACQUIRE) == g)
        __builtin_amdgcn_s_sleep(1);
    }
    __threadfence();                                        // acquire
  }
  __syncthreads();
}

__global__ void __launch_bounds__(256)
gru_persistent(const bf16* __restrict__ x0, bf16* __restrict__ x1,
               const bf16* __restrict__ wg, const bf16* __restrict__ whp,
               float* __restrict__ G, bf16* __restrict__ rh,
               float* __restrict__ hf, bf16* __restrict__ hb,
               const float* __restrict__ br, const float* __restrict__ bz,
               const float* __restrict__ bh,
               float* __restrict__ out, unsigned* __restrict__ sy) {
  extern __shared__ char smem[];
  unsigned* cnt = sy;
  unsigned* gen = sy + 1;

  const int lane  = threadIdx.x & 31;
  const int wave  = threadIdx.x >> 5;
  const int nl    = lane & 15;
  const int khalf = lane >> 4;
  const int mtile = wave & 3;
  const int lnt   = wave >> 2;                  // local ntile 0/1
  const int ntile = blockIdx.x * 2 + lnt;       // 0..95 (gates)
  const int m     = mtile * 16 + nl;            // batch row for A fragment
  const int col   = ntile * 16 + nl;            // gates output column

  // ---- stage weights into LDS (once) ----
  // gates: 4 chunks (lnt,layer) of 32KB at smem + c*32768
  for (int c = 0; c < 4; ++c) {
    int clnt = c >> 1, clayer = c & 1;
    const uint4* src = (const uint4*)(wg + (size_t)clayer * (KC * NG)
                                        + (size_t)(blockIdx.x * 2 + clnt) * 16384);
    uint4* dst = (uint4*)(smem + (size_t)c * 32768);
    for (int i = threadIdx.x; i < 2048; i += 256) dst[i] = src[i];
  }
  if (blockIdx.x < 16) {   // candidate GEMM weights: 4 chunks of 16KB at smem+131072
    for (int c = 0; c < 4; ++c) {
      int clnt = c >> 1, clayer = c & 1;
      const uint4* src = (const uint4*)(whp + (size_t)clayer * (GRU_H * GRU_H)
                                            + (size_t)(blockIdx.x * 2 + clnt) * 8192);
      uint4* dst = (uint4*)(smem + 131072 + (size_t)c * 16384);
      for (int i = threadIdx.x; i < 1024; i += 256) dst[i] = src[i];
    }
  }
  __syncthreads();

  // per-lane bias registers (output column fixed for whole sequence)
  const int nb = col & (GRU_H - 1);
  const float* bptr = (ntile < 32) ? br : ((ntile < 64) ? bz : bh);
  const float bias0 = bptr[nb], bias1 = bptr[GRU_H + nb];

  const int colh = (blockIdx.x * 2 + lnt) * 16 + nl;   // candidate column (blocks 0..15)

  for (int t = 0; t < GRU_T; ++t) {
    for (int l = 0; l < GRU_L; ++l) {
      float* hf_l = hf + (size_t)l * GRU_B * GRU_H;
      bf16*  hb_l = hb + (size_t)l * GRU_B * GRU_H;
      const float bias = l ? bias1 : bias0;

      // ---- phase A: G = [x_t,h_t] @ Wg  (B fragments from LDS), fused r*h ----
      {
        const bf16* xrow = (l == 0) ? (x0 + ((size_t)m * GRU_T + t) * GRU_H)
                                    : (x1 + ((size_t)t * GRU_B + m) * GRU_H);
        const bf16* hrow = hb_l + (size_t)m * GRU_H;
        const char* bbase = smem + (size_t)(lnt * 2 + l) * 32768;
        v8f acc = {};
#pragma unroll 4
        for (int kblk = 0; kblk < KC / 32; ++kblk) {
          const bf16* arow = (kblk < 16) ? (xrow + kblk * 32) : (hrow + (kblk - 16) * 32);
          v8bf alo = *(const v8bf*)(arow + khalf * 8);
          v8bf ahi = *(const v8bf*)(arow + 16 + khalf * 8);
          v16bf a = __builtin_shufflevector(alo, ahi, 0,1,2,3,4,5,6,7,8,9,10,11,12,13,14,15);
          v16bf b = *(const v16bf*)(bbase + (size_t)(((kblk * 2 + khalf) * 16 + nl) * 16) * 2);
          acc = __builtin_amdgcn_wmma_f32_16x16x32_bf16(false, a, false, b,
                                                        (short)0, acc, false, false);
        }
        if (ntile < 32) {            // r block: fold sigmoid + r*h here (tile-aligned)
#pragma unroll
          for (int v = 0; v < 8; ++v) {
            const int row = mtile * 16 + v + 8 * khalf;
            float r = sigmoidf_(acc[v] + bias);
            rh[(size_t)row * GRU_H + col] = (bf16)(r * hf_l[(size_t)row * GRU_H + col]);
          }
        } else {                     // z / h blocks: store preactivation with bias
#pragma unroll
          for (int v = 0; v < 8; ++v) {
            const int row = mtile * 16 + v + 8 * khalf;
            G[(size_t)row * NG + col] = acc[v] + bias;
          }
        }
      }
      grid_sync(cnt, gen);

      // ---- phase C: Hc = (r*h) @ Whh, fused gate math + stores (blocks 0..15) ----
      if (blockIdx.x < 16) {
        const bf16* arow0 = rh + (size_t)m * GRU_H;
        const char* bbase = smem + 131072 + (size_t)(lnt * 2 + l) * 16384;
        v8f acc = {};
#pragma unroll 4
        for (int kblk = 0; kblk < GRU_H / 32; ++kblk) {
          const bf16* arow = arow0 + kblk * 32;
          v8bf alo = *(const v8bf*)(arow + khalf * 8);
          v8bf ahi = *(const v8bf*)(arow + 16 + khalf * 8);
          v16bf a = __builtin_shufflevector(alo, ahi, 0,1,2,3,4,5,6,7,8,9,10,11,12,13,14,15);
          v16bf b = *(const v16bf*)(bbase + (size_t)(((kblk * 2 + khalf) * 16 + nl) * 16) * 2);
          acc = __builtin_amdgcn_wmma_f32_16x16x32_bf16(false, a, false, b,
                                                        (short)0, acc, false, false);
        }
#pragma unroll
        for (int v = 0; v < 8; ++v) {
          const int row = mtile * 16 + v + 8 * khalf;
          const float* grow = G + (size_t)row * NG;
          float z  = sigmoidf_(grow[GRU_H + colh]);          // bias pre-added in phase A
          float ht = tanhf(grow[2 * GRU_H + colh] + acc[v]); // bias pre-added in phase A
          float hx = hf_l[(size_t)row * GRU_H + colh];
          float hy = z * hx + (1.0f - z) * ht;
          hf_l[(size_t)row * GRU_H + colh] = hy;
          hb_l[(size_t)row * GRU_H + colh] = (bf16)hy;
          if (l == 0) x1[((size_t)t * GRU_B + row) * GRU_H + colh] = (bf16)hy;
          else        out[((size_t)row * GRU_T + t) * GRU_H + colh] = hy;
        }
      }
      grid_sync(cnt, gen);
    }
  }

  // ---- epilogue: hidden_final (B,L,H) ----
  float* outh = out + (size_t)GRU_B * GRU_T * GRU_H;
  for (int idx = blockIdx.x * 256 + threadIdx.x; idx < GRU_L * GRU_B * GRU_H;
       idx += NBLK * 256) {
    int n = idx & (GRU_H - 1);
    int b = (idx >> 9) & (GRU_B - 1);
    int l = idx >> 15;
    outh[((size_t)b * GRU_L + l) * GRU_H + n] = hf[idx];
  }
}

// ---------------- host ----------------

extern "C" void kernel_launch(void* const* d_in, const int* in_sizes, int n_in,
                              void* d_out, int out_size, void* d_ws, size_t ws_size,
                              hipStream_t stream) {
  const float* input  = (const float*)d_in[0];
  const float* hidden = (const float*)d_in[1];
  const float* Wxz    = (const float*)d_in[2];
  const float* Whz    = (const float*)d_in[3];
  const float* bz     = (const float*)d_in[4];
  const float* Wxr    = (const float*)d_in[5];
  const float* Whr    = (const float*)d_in[6];
  const float* br     = (const float*)d_in[7];
  const float* Wxh    = (const float*)d_in[8];
  const float* Whh    = (const float*)d_in[9];
  const float* bh     = (const float*)d_in[10];
  float* out = (float*)d_out;

  char* ws = (char*)d_ws;
  bf16*     x0 = (bf16*)(ws + OFF_X0);
  bf16*     x1 = (bf16*)(ws + OFF_X1);
  bf16*     wg = (bf16*)(ws + OFF_WG);
  bf16*     wh = (bf16*)(ws + OFF_WH);
  float*    G  = (float*)(ws + OFF_G);
  bf16*     rh = (bf16*)(ws + OFF_RH);
  float*    hf = (float*)(ws + OFF_HF);
  bf16*     hb = (bf16*)(ws + OFF_HB);
  unsigned* sy = (unsigned*)(ws + OFF_SY);

  (void)hipFuncSetAttribute((const void*)gru_persistent,
                            hipFuncAttributeMaxDynamicSharedMemorySize, LDS_BYTES);

  // prologue: convert input, init state/sync, pack weights into WMMA fragment order
  gru_cvt_in  <<<4096, 256, 0, stream>>>(input, x0, GRU_B * GRU_T * GRU_H);
  gru_init_h  <<<(GRU_L * GRU_B * GRU_H) / 256, 256, 0, stream>>>(hidden, hf, hb);
  gru_zero_sync<<<1, 64, 0, stream>>>(sy);
  gru_pack_wg <<<(GRU_L * KC * NG) / 256, 256, 0, stream>>>(Wxr, Whr, Wxz, Whz, Wxh, wg);
  gru_pack_wh <<<(GRU_L * GRU_H * GRU_H) / 256, 256, 0, stream>>>(Whh, wh);

  // one persistent kernel runs the whole 1024-step, 2-layer recurrence:
  // weights LDS-resident, grid barriers instead of kernel launches
  gru_persistent<<<NBLK, 256, LDS_BYTES, stream>>>(x0, x1, wg, wh, G, rh, hf, hb,
                                                   br, bz, bh, out, sy);

  (void)in_sizes; (void)n_in; (void)out_size; (void)ws_size;
}
